// AttentionNeuralOperator_29171417874655
// MI455X (gfx1250) — matrix operand order repaired and checked by reading, not compiled, not run on hardware
//
#include <hip/hip_runtime.h>

// ---------------------------------------------------------------------------
// Shapes (fixed by the reference): b=2,s=32 -> BS=64; c=128; h=w=24 -> L=576
// heads=2, qk head dim=64, hid=256 (v head dim=128), out=128.
// ---------------------------------------------------------------------------
#define BS   64
#define CIN  128
#define LL   576
#define NH   2
#define QKD  64
#define HID  256
#define VD   128
#define OUTC 128

typedef __attribute__((ext_vector_type(16))) _Float16 v16h;
typedef __attribute__((ext_vector_type(8)))  float    v8f;
typedef unsigned int u32x4 __attribute__((ext_vector_type(4)));
typedef int          i32x4 __attribute__((ext_vector_type(4)));
typedef int          i32x8 __attribute__((ext_vector_type(8)));

#if __has_builtin(__builtin_amdgcn_tensor_load_to_lds)
#define USE_TDM 1
#endif

struct alignas(16) H8 { _Float16 h[8]; };

// A fragment (16x32, f16): lane&15 = row M; lanes<16 hold K = {0..7,16..23},
// lanes>=16 hold +8.  rowp points at &src[M][k0] (K contiguous in memory).
static __device__ inline v16h frag_a_f16(const _Float16* rowp, int hi) {
  const _Float16* p = rowp + 8 * hi;
  H8 lo = *(const H8*)(p);
  H8 hh = *(const H8*)(p + 16);
  v16h a;
#pragma unroll
  for (int e = 0; e < 8; ++e) { a[e] = lo.h[e]; a[8 + e] = hh.h[e]; }
  return a;
}

// B fragment (32x16, f16): lane&15 = column N; lanes<16 hold K = {0..7,16..23},
// lanes>=16 hold +8.  colp points at &srcT[N][k0] (K contiguous).
static __device__ inline v16h frag_b_f16(const _Float16* colp, int hi) {
  const _Float16* p = colp + 8 * hi;
  H8 lo = *(const H8*)(p);
  H8 hh = *(const H8*)(p + 16);
  v16h b;
#pragma unroll
  for (int e = 0; e < 8; ++e) { b[e] = lo.h[e]; b[8 + e] = hh.h[e]; }
  return b;
}

static __device__ inline v8f wmma16(v16h a, v16h b, v8f c) {
  // v_wmma_f32_16x16x32_f16  (neg_a, A, neg_b, B, c_mod, C, reuse_a, reuse_b)
  return __builtin_amdgcn_wmma_f32_16x16x32_f16(false, a, false, b, (short)0, c,
                                                false, false);
}

// ---------------------------------------------------------------------------
// K0: pack weights to f16. Wcat = [qk_w ; v_w] (512x128), W1h (256x256),
// W2h (128x256).
// ---------------------------------------------------------------------------
__global__ __launch_bounds__(256) void prep_weights_kernel(
    const float* qk_w, const float* v_w, const float* w1, const float* w2,
    _Float16* Wcat, _Float16* W1h, _Float16* W2h) {
  int t = blockIdx.x * 256 + threadIdx.x;
  if (t < 65536) {
    float v = (t < 32768) ? qk_w[t] : v_w[t - 32768];
    Wcat[t] = (_Float16)v;
  } else if (t < 131072) {
    W1h[t - 65536] = (_Float16)w1[t - 65536];
  } else if (t < 163840) {
    W2h[t - 131072] = (_Float16)w2[t - 131072];
  }
}

// ---------------------------------------------------------------------------
// K1: Swin-v2 continuous relative position bias.  biasW[head][l][m], f32.
// ---------------------------------------------------------------------------
__global__ __launch_bounds__(256) void cpb_bias_kernel(
    const float* w1, const float* b1, const float* w2, float* biasW) {
  __shared__ float sW1[256];   // (128,2)
  __shared__ float sB1[128];
  __shared__ float sW2[256];   // (2,128)
  int t = threadIdx.x;
  sW1[t] = w1[t];
  sW2[t] = w2[t];
  if (t < 128) sB1[t] = b1[t];
  __syncthreads();

  int pair = blockIdx.x * 256 + t;           // 1296*256 == 576*576 exactly
  int i = pair / LL, j = pair % LL;
  int yi = i / 24, xi = i % 24, yj = j / 24, xj = j % 24;
  float r0 = (float)(yi - yj) * (8.0f / 23.0f);
  float r1 = (float)(xi - xj) * (8.0f / 23.0f);
  r0 = copysignf(log2f(1.0f + fabsf(r0)) * (1.0f / 3.0f), r0);  // log2(8)=3
  r1 = copysignf(log2f(1.0f + fabsf(r1)) * (1.0f / 3.0f), r1);
  float a0 = 0.0f, a1 = 0.0f;
#pragma unroll 4
  for (int k = 0; k < 128; ++k) {
    float hv = fmaf(sW1[2 * k], r0, fmaf(sW1[2 * k + 1], r1, sB1[k]));
    hv = fmaxf(hv, 0.0f);
    a0 = fmaf(sW2[k], hv, a0);
    a1 = fmaf(sW2[128 + k], hv, a1);
  }
  biasW[pair] = a0;
  biasW[(size_t)LL * LL + pair] = a1;
}

// ---------------------------------------------------------------------------
// K2: QKV projection GEMM (512 x 576 x 128 per image).
// ---------------------------------------------------------------------------
__global__ __launch_bounds__(256) void proj_kernel(
    const float* x, const _Float16* Wcat,
    _Float16* Qh, _Float16* Kh, _Float16* Vt) {
  __shared__ _Float16 XT[64][40];  // x tile transposed: [l][c], stride 40

  int tid = threadIdx.x, lane = tid & 31, wave = tid >> 5;
  int hi = lane >> 4, ln = lane & 15;
  int bsb = blockIdx.x / 36;
  int rem = blockIdx.x % 36;
  int Mb = (rem / 9) * 128;
  int Nb = (rem % 9) * 64;
  int wm = wave >> 1, wn = wave & 1;
  int M0 = Mb + wm * 32;
  int N0w = wn * 32;

  v8f acc[2][2] = {};
  const float* xb = x + (size_t)bsb * CIN * LL;

  for (int c0 = 0; c0 < 128; c0 += 32) {
    int idx = tid * 8;
    int crel = idx >> 6;        // 0..31
    int lrel = idx & 63;        // multiple of 8
    const float* src = xb + (size_t)(c0 + crel) * LL + Nb + lrel;
    float4 f0 = *(const float4*)src;
    float4 f1 = *(const float4*)(src + 4);
    XT[lrel + 0][crel] = (_Float16)f0.x;
    XT[lrel + 1][crel] = (_Float16)f0.y;
    XT[lrel + 2][crel] = (_Float16)f0.z;
    XT[lrel + 3][crel] = (_Float16)f0.w;
    XT[lrel + 4][crel] = (_Float16)f1.x;
    XT[lrel + 5][crel] = (_Float16)f1.y;
    XT[lrel + 6][crel] = (_Float16)f1.z;
    XT[lrel + 7][crel] = (_Float16)f1.w;
    __syncthreads();

    v16h a0 = frag_a_f16(Wcat + (size_t)(M0 + ln) * 128 + c0, hi);
    v16h a1 = frag_a_f16(Wcat + (size_t)(M0 + 16 + ln) * 128 + c0, hi);
    v16h b0 = frag_b_f16(&XT[N0w + ln][0], hi);
    v16h b1 = frag_b_f16(&XT[N0w + 16 + ln][0], hi);
    acc[0][0] = wmma16(a0, b0, acc[0][0]);
    acc[0][1] = wmma16(a0, b1, acc[0][1]);
    acc[1][0] = wmma16(a1, b0, acc[1][0]);
    acc[1][1] = wmma16(a1, b1, acc[1][1]);
    __syncthreads();
  }

#pragma unroll
  for (int i = 0; i < 2; ++i)
#pragma unroll
    for (int j = 0; j < 2; ++j)
#pragma unroll
      for (int r = 0; r < 8; ++r) {
        int m = 8 * hi + r;
        int og = M0 + i * 16 + m;                // 0..511
        int lg = Nb + N0w + j * 16 + ln;         // 0..575
        _Float16 hv = (_Float16)acc[i][j][r];
        if (og < 256) {                          // qk: o = ((sel*2+head)*64+d)
          int sel = og >> 7, head = (og >> 6) & 1, d = og & 63;
          _Float16* dst = sel ? Kh : Qh;
          dst[(((size_t)bsb * 2 + head) * LL + lg) * QKD + d] = hv;
        } else {                                 // v: head*128 + dv
          int ov = og - 256, head = ov >> 7, dv = ov & 127;
          Vt[(((size_t)bsb * 2 + head) * VD + dv) * LL + lg] = hv;
        }
      }
}

// ---------------------------------------------------------------------------
// K3: attention.  One block per (bs, head, 16 query rows).
// The 16x64 Q tile is DMA'd into LDS once via the Tensor Data Mover
// (TENSOR_LOAD_TO_LDS + s_wait_tensorcnt) and reused by all 8 waves.
// ---------------------------------------------------------------------------
__global__ __launch_bounds__(256) void attn_kernel(
    const _Float16* Qh, const _Float16* Kh, const _Float16* Vt,
    const float* biasW, const float* sa_bias, _Float16* Sh) {
  __shared__ float sc[16 * LL];                 // 36 KB score strip
  __shared__ float red[16][17];
  __shared__ alignas(16) _Float16 qtile[16][QKD];  // 2 KB Q tile

  int tid = threadIdx.x, lane = tid & 31, wave = tid >> 5;
  int hi = lane >> 4, ln = lane & 15;
  int bsb = blockIdx.x / (NH * 36);
  int r2 = blockIdx.x % (NH * 36);
  int head = r2 / 36;
  int qb = (r2 % 36) * 16;
  size_t qkbase = ((size_t)bsb * 2 + head) * LL;
  const float scale = 0.125f;  // 1/sqrt(64)

  // ---- stage Q tile (16 rows x 64 halfs) into LDS ----
#ifdef USE_TDM
  if (wave == 0) {
    // D# per CDNA5 ISA (2D tile, data_size=2B)
    unsigned lds_off = (unsigned)(unsigned long long)&qtile[0][0];  // addr[31:0]
    unsigned long long ga =
        (unsigned long long)(const void*)(Qh + (qkbase + qb) * QKD);
    u32x4 g0;
    g0[0] = 1u;                                   // count=1, user descriptor
    g0[1] = lds_off;                              // lds_addr (bytes)
    g0[2] = (unsigned)ga;                         // global_addr[31:0]
    g0[3] = (unsigned)((ga >> 32) & 0x1FFFFFFu)   // global_addr[56:32]
            | (2u << 30);                         // type=2 ("image")
    i32x8 g1;
    g1[0] = 1 << 16;          // workgroup_mask=0, data_size=1 (2 bytes)
    g1[1] = QKD << 16;        // tensor_dim0[15:0]=64 (bits 63:48 of group)
    g1[2] = 16 << 16;         // tensor_dim0 hi=0, tensor_dim1[15:0]=16
    g1[3] = QKD << 16;        // tensor_dim1 hi=0, tile_dim0=64
    g1[4] = 16;               // tile_dim1=16, tile_dim2=0
    g1[5] = QKD;              // tensor_dim0_stride low32 = 64 elements
    g1[6] = 0;                // stride hi / tensor_dim1_stride low
    g1[7] = 0;
    i32x4 g2 = {0, 0, 0, 0};
    i32x4 g3 = {0, 0, 0, 0};
#if __clang_major__ >= 23
    i32x8 g4 = {0, 0, 0, 0, 0, 0, 0, 0};
    __builtin_amdgcn_tensor_load_to_lds(g0, g1, g2, g3, g4, 0);
#else
    __builtin_amdgcn_tensor_load_to_lds(g0, g1, g2, g3, 0);
#endif
    __builtin_amdgcn_s_wait_tensorcnt((short)0);
  }
#else
  {
    int idx = tid * 4;                            // 256 thr * 4 halfs = 1024
    int row = idx >> 6, col = idx & 63;
    *(unsigned long long*)&qtile[row][col] =
        *(const unsigned long long*)(Qh + (qkbase + qb + row) * QKD + col);
  }
#endif
  __syncthreads();

  // ---- phase 1: scores = Q K^T * scale + bias ----
  for (int nt = wave; nt < 36; nt += 8) {
    v8f acc = {};
#pragma unroll
    for (int c0 = 0; c0 < 64; c0 += 32) {
      v16h a = frag_a_f16(&qtile[ln][c0], hi);
      v16h b = frag_b_f16(Kh + (qkbase + nt * 16 + ln) * QKD + c0, hi);
      acc = wmma16(a, b, acc);
    }
#pragma unroll
    for (int r = 0; r < 8; ++r) {
      int m = 8 * hi + r;
      int keyg = nt * 16 + ln;
      float bv = biasW[(size_t)head * LL * LL + (size_t)(qb + m) * LL + keyg];
      sc[m * LL + keyg] = acc[r] * scale + bv;
    }
  }
  __syncthreads();

  // ---- phase 2: softmax over rows of sc ----
  int row = tid >> 4, sub = tid & 15;
  float mx = -3.0e38f;
  for (int j = sub; j < LL; j += 16) mx = fmaxf(mx, sc[row * LL + j]);
  red[row][sub] = mx;
  __syncthreads();
  float rmax = -3.0e38f;
#pragma unroll
  for (int tt = 0; tt < 16; ++tt) rmax = fmaxf(rmax, red[row][tt]);
  float s = 0.0f;
  for (int j = sub; j < LL; j += 16) {
    float p = __expf(sc[row * LL + j] - rmax);
    sc[row * LL + j] = p;
    s += p;
  }
  __syncthreads();
  red[row][sub] = s;
  __syncthreads();
  float rs = 0.0f;
#pragma unroll
  for (int tt = 0; tt < 16; ++tt) rs += red[row][tt];
  float inv = 1.0f / rs;
  for (int j = sub; j < LL; j += 16) sc[row * LL + j] *= inv;
  __syncthreads();

  // ---- phase 3: P @ V ----
  int dvb = wave * 16;
  v8f acc = {};
  const _Float16* vb = Vt + (((size_t)bsb * 2 + head) * VD + dvb + ln) * LL;
  for (int kb = 0; kb < LL; kb += 32) {
    v16h a;
#pragma unroll
    for (int e = 0; e < 16; ++e) {
      int kk = kb + 8 * hi + (e < 8 ? e : e + 8);
      a[e] = (_Float16)sc[ln * LL + kk];
    }
    v16h b = frag_b_f16(vb + kb, hi);
    acc = wmma16(a, b, acc);
  }
  int ch = head * VD + dvb + ln;
  float sbv = sa_bias[ch];
#pragma unroll
  for (int r = 0; r < 8; ++r) {
    int m = 8 * hi + r;
    int lq = qb + m;
    Sh[((size_t)bsb * LL + lq) * HID + ch] = (_Float16)(acc[r] + sbv);
  }
}

// ---------------------------------------------------------------------------
// K4: MLP layer 1 (36864 x 256, K=256) + exact GELU.
// ---------------------------------------------------------------------------
__global__ __launch_bounds__(256) void mlp1_kernel(
    const _Float16* Sh, const _Float16* W1h, const float* b1, _Float16* Hh) {
  int tid = threadIdx.x, lane = tid & 31, wave = tid >> 5;
  int hi = lane >> 4, ln = lane & 15;
  int br = blockIdx.x >> 2, bc = blockIdx.x & 3;
  int wm = wave >> 1, wn = wave & 1;
  int M0 = br * 64 + wm * 16;
  int N0 = bc * 64 + wn * 32;

  v8f acc[2] = {};
  const _Float16* arow = Sh + (size_t)(M0 + ln) * HID;
  for (int k0 = 0; k0 < HID; k0 += 32) {
    __builtin_prefetch(arow + k0 + 64, 0, 1);   // -> global_prefetch_b8
    v16h a = frag_a_f16(arow + k0, hi);
    v16h b0 = frag_b_f16(W1h + (size_t)(N0 + ln) * HID + k0, hi);
    v16h b1v = frag_b_f16(W1h + (size_t)(N0 + 16 + ln) * HID + k0, hi);
    acc[0] = wmma16(a, b0, acc[0]);
    acc[1] = wmma16(a, b1v, acc[1]);
  }
#pragma unroll
  for (int j = 0; j < 2; ++j) {
    int col = N0 + j * 16 + ln;
    float bb = b1[col];
#pragma unroll
    for (int r = 0; r < 8; ++r) {
      int m = 8 * hi + r;
      float v = acc[j][r] + bb;
      float g = 0.5f * v * (1.0f + erff(v * 0.70710678118654752f));
      Hh[(size_t)(M0 + m) * HID + col] = (_Float16)g;
    }
  }
}

// ---------------------------------------------------------------------------
// K5: MLP layer 2 (36864 x 128, K=256) -> f32 output in (b,s,out,h,w) layout.
// ---------------------------------------------------------------------------
__global__ __launch_bounds__(256) void mlp2_kernel(
    const _Float16* Hh, const _Float16* W2h, const float* b2, float* out) {
  int tid = threadIdx.x, lane = tid & 31, wave = tid >> 5;
  int hi = lane >> 4, ln = lane & 15;
  int br = blockIdx.x >> 1, bc = blockIdx.x & 1;
  int wm = wave >> 1, wn = wave & 1;
  int M0 = br * 64 + wm * 16;
  int N0 = bc * 64 + wn * 32;

  v8f acc[2] = {};
  const _Float16* arow = Hh + (size_t)(M0 + ln) * HID;
  for (int k0 = 0; k0 < HID; k0 += 32) {
    v16h a = frag_a_f16(arow + k0, hi);
    v16h b0 = frag_b_f16(W2h + (size_t)(N0 + ln) * HID + k0, hi);
    v16h b1v = frag_b_f16(W2h + (size_t)(N0 + 16 + ln) * HID + k0, hi);
    acc[0] = wmma16(a, b0, acc[0]);
    acc[1] = wmma16(a, b1v, acc[1]);
  }
#pragma unroll
  for (int j = 0; j < 2; ++j) {
    int col = N0 + j * 16 + ln;
    float bb = b2[col];
#pragma unroll
    for (int r = 0; r < 8; ++r) {
      int m = 8 * hi + r;
      int p = M0 + m;                 // global pixel = bs*576 + l
      int bsb = p / LL, lq = p % LL;
      out[((size_t)bsb * OUTC + col) * LL + lq] = acc[j][r] + bb;
    }
  }
}

// ---------------------------------------------------------------------------
extern "C" void kernel_launch(void* const* d_in, const int* in_sizes, int n_in,
                              void* d_out, int out_size, void* d_ws,
                              size_t ws_size, hipStream_t stream) {
  const float* x       = (const float*)d_in[0];
  const float* qk_w    = (const float*)d_in[1];
  const float* v_w     = (const float*)d_in[2];
  const float* cpb_w1  = (const float*)d_in[3];
  const float* cpb_b1  = (const float*)d_in[4];
  const float* cpb_w2  = (const float*)d_in[5];
  const float* sa_bias = (const float*)d_in[6];
  const float* mlp_w1  = (const float*)d_in[7];
  const float* mlp_b1  = (const float*)d_in[8];
  const float* mlp_w2  = (const float*)d_in[9];
  const float* mlp_b2  = (const float*)d_in[10];
  float* out = (float*)d_out;

  char* ws = (char*)d_ws;
  size_t off = 0;
  auto walloc = [&](size_t bytes) -> void* {
    void* p = (void*)(ws + off);
    off = (off + bytes + 255) & ~(size_t)255;
    return p;
  };
  float*    biasW = (float*)walloc((size_t)NH * LL * LL * 4);              // 2.65 MB
  _Float16* Qh    = (_Float16*)walloc((size_t)BS * NH * LL * QKD * 2);     // 9.4 MB
  _Float16* Kh    = (_Float16*)walloc((size_t)BS * NH * LL * QKD * 2);     // 9.4 MB
  _Float16* Vt    = (_Float16*)walloc((size_t)BS * NH * VD * LL * 2);      // 18.9 MB
  _Float16* Sh    = (_Float16*)walloc((size_t)BS * LL * HID * 2);          // 18.9 MB
  _Float16* Hh    = (_Float16*)walloc((size_t)BS * LL * HID * 2);          // 18.9 MB
  _Float16* Wcat  = (_Float16*)walloc(512 * 128 * 2);
  _Float16* W1h   = (_Float16*)walloc(256 * 256 * 2);
  _Float16* W2h   = (_Float16*)walloc(128 * 256 * 2);
  (void)in_sizes; (void)n_in; (void)out_size; (void)ws_size;

  prep_weights_kernel<<<640, 256, 0, stream>>>(qk_w, v_w, mlp_w1, mlp_w2,
                                               Wcat, W1h, W2h);
  cpb_bias_kernel<<<1296, 256, 0, stream>>>(cpb_w1, cpb_b1, cpb_w2, biasW);
  proj_kernel<<<BS * 4 * 9, 256, 0, stream>>>(x, Wcat, Qh, Kh, Vt);
  attn_kernel<<<BS * NH * 36, 256, 0, stream>>>(Qh, Kh, Vt, biasW, sa_bias, Sh);
  mlp1_kernel<<<(BS * LL / 64) * (HID / 64), 256, 0, stream>>>(Sh, W1h, mlp_b1, Hh);
  mlp2_kernel<<<(BS * LL / 64) * (OUTC / 64), 256, 0, stream>>>(Hh, W2h, mlp_b2, out);
}
